// MultiHeadedAttention_21311627722912
// MI455X (gfx1250) — compile-verified, matrix operand order checked by reading, not run on hardware
//
#include <hip/hip_runtime.h>
#include <hip/hip_bf16.h>

typedef __attribute__((ext_vector_type(16))) _Float16 v16h;
typedef __attribute__((ext_vector_type(8)))  _Float16 v8h;
typedef __attribute__((ext_vector_type(8)))  float    v8f;
typedef __attribute__((ext_vector_type(4)))  float    v4f;

union Frag16 { v16h v; v8h h[2]; };

#define D_MODEL 1024
#define HEADS   16
#define DK      64
#define SEQ     2048

// Async global->LDS copy of 16 bytes (CDNA5 path, ASYNCcnt-tracked).
#define ASYNC_B128(ldsOff, gptr)                                   \
  asm volatile("global_load_async_to_lds_b128 %0, %1, off"         \
               :: "v"(ldsOff), "v"(gptr) : "memory")

// ---------------------------------------------------------------------------
// GEMM: Out[M x 1024] = X[M x 1024] * W[1024 x 1024] + bias
//   X_HALF:   X is f16 (else fp32)
//   OUT_HEAD: store f16 head-split [B,H,S,DK]; else fp32 row-major [M,1024]
// Block: 256 threads (8 waves). Block tile 128(M) x 64(N), K-step 32.
// Register-buffered pipeline: global loads of tile kc+1 overlap WMMA of kc.
// ---------------------------------------------------------------------------
template<bool X_HALF, bool OUT_HEAD>
__global__ void __launch_bounds__(256) gemm_wmma(
    const void* __restrict__ Xv, const float* __restrict__ W,
    const float* __restrict__ bias, void* __restrict__ Outv)
{
  __shared__ _Float16 Alds[128][40];   // 80B row stride (16B aligned)
  __shared__ _Float16 Btl[64][40];     // W tile transposed: Btl[n][k]

  const int tid  = threadIdx.x;
  const int wave = tid >> 5;
  const int lane = tid & 31;
  const int ln   = lane & 15;
  const int hi   = lane >> 4;

  const int n0 = blockIdx.x * 64;
  const int m0 = blockIdx.y * 128;

  const float*    Xf = (const float*)Xv;
  const _Float16* Xh = (const _Float16*)Xv;

  v8f acc[4] = {};

  const int arow  = tid >> 1;          // 0..127
  const int akofs = (tid & 1) * 16;    // 0 or 16
  const int bk    = tid >> 3;          // 0..31
  const int bn    = (tid & 7) * 8;     // 0..56

  // staging registers for the next K-step
  v4f ra[4];           // X tile (fp32 path)
  v8h rah[2];          // X tile (f16 path)
  v4f rb[2];           // W tile

  auto loadRegs = [&](int kc) {
    if (X_HALF) {
      const _Float16* src = Xh + (size_t)(m0 + arow) * D_MODEL + kc * 32 + akofs;
      rah[0] = *(const v8h*)(src);
      rah[1] = *(const v8h*)(src + 8);
    } else {
      const float* src = Xf + (size_t)(m0 + arow) * D_MODEL + kc * 32 + akofs;
      ra[0] = *(const v4f*)(src);
      ra[1] = *(const v4f*)(src + 4);
      ra[2] = *(const v4f*)(src + 8);
      ra[3] = *(const v4f*)(src + 12);
    }
    const float* wsrc = W + (size_t)(kc * 32 + bk) * D_MODEL + n0 + bn;
    rb[0] = *(const v4f*)(wsrc);
    rb[1] = *(const v4f*)(wsrc + 4);
  };

  loadRegs(0);

  for (int kc = 0; kc < D_MODEL / 32; ++kc) {
    __syncthreads();                   // LDS free (everyone done with kc-1)
    // ---- commit staged tile kc to LDS (f16) ----
    if (X_HALF) {
      *(v8h*)&Alds[arow][akofs]     = rah[0];
      *(v8h*)&Alds[arow][akofs + 8] = rah[1];
    } else {
      v8h h0, h1;
      #pragma unroll
      for (int i = 0; i < 4; ++i) {
        h0[i]     = (_Float16)ra[0][i]; h0[i + 4] = (_Float16)ra[1][i];
        h1[i]     = (_Float16)ra[2][i]; h1[i + 4] = (_Float16)ra[3][i];
      }
      *(v8h*)&Alds[arow][akofs]     = h0;
      *(v8h*)&Alds[arow][akofs + 8] = h1;
    }
    {
      #pragma unroll
      for (int i = 0; i < 4; ++i) Btl[bn + i][bk]     = (_Float16)rb[0][i];
      #pragma unroll
      for (int i = 0; i < 4; ++i) Btl[bn + 4 + i][bk] = (_Float16)rb[1][i];
    }
    // ---- start global loads for kc+1 (overlap with WMMA below) ----
    if (kc + 1 < D_MODEL / 32) loadRegs(kc + 1);
    __syncthreads();                   // tile kc visible

    // ---- per-wave WMMA ----
    Frag16 a;
    {
      const int r = wave * 16 + ln;                 // A row M = lane&15
      a.h[0] = *(const v8h*)&Alds[r][hi * 8];       // K = hi*8 .. +7
      a.h[1] = *(const v8h*)&Alds[r][16 + hi * 8];  // K = 16+hi*8 .. +7
    }
    #pragma unroll
    for (int nb = 0; nb < 4; ++nb) {
      Frag16 bfr;
      const int r = nb * 16 + ln;                   // B col N = lane&15
      bfr.h[0] = *(const v8h*)&Btl[r][hi * 8];
      bfr.h[1] = *(const v8h*)&Btl[r][16 + hi * 8];
      acc[nb] = __builtin_amdgcn_wmma_f32_16x16x32_f16(
          false, a.v, false, bfr.v, (short)0, acc[nb], false, false);
    }
  }

  // ---- epilogue: bias + store ----
  #pragma unroll
  for (int nb = 0; nb < 4; ++nb) {
    const int n = n0 + nb * 16 + ln;
    const float bv = bias[n];
    #pragma unroll
    for (int j = 0; j < 8; ++j) {
      const int m = m0 + wave * 16 + j + 8 * hi;    // C row M = j + 8*hi
      const float val = acc[nb][j] + bv;
      if (OUT_HEAD) {
        const int bidx = m >> 11;                   // m / SEQ
        const int s    = m & (SEQ - 1);
        const int h    = n >> 6;                    // n / DK
        const int dk   = n & 63;
        ((_Float16*)Outv)[(((size_t)(bidx * HEADS + h)) * SEQ + s) * DK + dk] =
            (_Float16)val;
      } else {
        ((float*)Outv)[(size_t)m * D_MODEL + n] = val;
      }
    }
  }
}

// ---------------------------------------------------------------------------
// Flash attention: one block per (b, h, 128-row q tile). 8 waves x 16 q rows.
// kv chunk = 64, double-buffered: chunk t+1 staged (K via async global->LDS,
// V transposed via VALU+ds) while chunk t runs on the WMMA pipes. Row sums
// accumulate via WMMA against a ones matrix (no sum shuffles).
// ---------------------------------------------------------------------------
__global__ void __launch_bounds__(256) attn_wmma(
    const _Float16* __restrict__ Q, const _Float16* __restrict__ K,
    const _Float16* __restrict__ V, _Float16* __restrict__ Out)
{
  __shared__ _Float16 Klds[2][64][72];   // K chunk [kv][dk], 144B row stride
  __shared__ _Float16 Vt[2][64][72];     // V chunk transposed [dk][kv]
  __shared__ _Float16 Plds[8][16][72];   // per-wave P tile [qrow][kv]

  const int tid  = threadIdx.x;
  const int wave = tid >> 5;
  const int lane = tid & 31;
  const int ln   = lane & 15;
  const int hi   = lane >> 4;

  const int qt = blockIdx.x;
  const int h  = blockIdx.y;
  const int b  = blockIdx.z;

  const _Float16* Qh = Q + (((size_t)b * HEADS + h) * SEQ) * DK;
  const _Float16* Kh = K + (((size_t)b * HEADS + h) * SEQ) * DK;
  const _Float16* Vh = V + (((size_t)b * HEADS + h) * SEQ) * DK;

  const int q0 = qt * 128 + wave * 16;

  // Q fragments for dk chunks {0..31} and {32..63}
  Frag16 aq[2];
  {
    const _Float16* qr = Qh + (size_t)(q0 + ln) * DK;
    aq[0].h[0] = *(const v8h*)(qr + hi * 8);
    aq[0].h[1] = *(const v8h*)(qr + 16 + hi * 8);
    aq[1].h[0] = *(const v8h*)(qr + 32 + hi * 8);
    aq[1].h[1] = *(const v8h*)(qr + 48 + hi * 8);
  }

  // B-matrix of all ones (for l = P x ones row-sum accumulation)
  Frag16 ones;
  #pragma unroll
  for (int i = 0; i < 8; ++i) {
    ones.h[0][i] = (_Float16)1.0f;
    ones.h[1][i] = (_Float16)1.0f;
  }

  v8f o[4] = {};
  v8f ol   = {};      // running softmax denominators (all 16 cols identical)
  float mrow[8];
  #pragma unroll
  for (int j = 0; j < 8; ++j) mrow[j] = -3.0e38f;

  const int kr = tid >> 2;         // 0..63  (kv row)
  const int kq = (tid & 3) * 16;   // 0,16,32,48 (dk offset, 16 halves/thread)

  // Stage chunk t into buffer buf: K async -> LDS, V load+transpose -> LDS.
  auto stage = [&](int t, int buf) {
    const int kv0 = t * 64;
    {
      const _Float16* gp = Kh + (size_t)(kv0 + kr) * DK + kq;
      unsigned l0 = (unsigned)(uintptr_t)&Klds[buf][kr][kq];
      unsigned l1 = (unsigned)(uintptr_t)&Klds[buf][kr][kq + 8];
      ASYNC_B128(l0, gp);
      ASYNC_B128(l1, gp + 8);
      __builtin_prefetch(gp + 64 * DK, 0, 0);      // chunk t+1 into L2
    }
    {
      const _Float16* vp = Vh + (size_t)(kv0 + kr) * DK + kq;
      __builtin_prefetch(vp + 64 * DK, 0, 0);
      v8h d0 = *(const v8h*)(vp);
      v8h d1 = *(const v8h*)(vp + 8);
      #pragma unroll
      for (int i = 0; i < 8; ++i) Vt[buf][kq + i][kr]     = d0[i];
      #pragma unroll
      for (int i = 0; i < 8; ++i) Vt[buf][kq + 8 + i][kr] = d1[i];
    }
  };

  stage(0, 0);

  for (int t = 0; t < SEQ / 64; ++t) {
    const int cur = t & 1;
    asm volatile("s_wait_asynccnt 0" ::: "memory");
    __syncthreads();   // buffer 'cur' fully staged & prior compute retired

    if (t + 1 < SEQ / 64) stage(t + 1, cur ^ 1);   // overlap with compute

    // ---- scores S = Q * K^T : 16 x 64, four 16x16 accs, reduce dk=64 ----
    v8f s[4] = {};
    #pragma unroll
    for (int c = 0; c < 2; ++c) {
      #pragma unroll
      for (int nb = 0; nb < 4; ++nb) {
        Frag16 bfr;
        const int r = nb * 16 + ln;              // kv column
        bfr.h[0] = *(const v8h*)&Klds[cur][r][c * 32 + hi * 8];
        bfr.h[1] = *(const v8h*)&Klds[cur][r][c * 32 + 16 + hi * 8];
        s[nb] = __builtin_amdgcn_wmma_f32_16x16x32_f16(
            false, aq[c].v, false, bfr.v, (short)0, s[nb], false, false);
      }
    }

    // ---- online softmax: max via 4 shuffles/row; sums via WMMA below ----
    #pragma unroll
    for (int j = 0; j < 8; ++j) {
      s[0][j] *= 0.125f; s[1][j] *= 0.125f;      // 1/sqrt(64)
      s[2][j] *= 0.125f; s[3][j] *= 0.125f;
      float cm = fmaxf(fmaxf(s[0][j], s[1][j]), fmaxf(s[2][j], s[3][j]));
      cm = fmaxf(cm, __shfl_xor(cm, 1, 32));
      cm = fmaxf(cm, __shfl_xor(cm, 2, 32));
      cm = fmaxf(cm, __shfl_xor(cm, 4, 32));
      cm = fmaxf(cm, __shfl_xor(cm, 8, 32));
      const float mnew = fmaxf(mrow[j], cm);
      const float corr = __expf(mrow[j] - mnew);
      mrow[j] = mnew;
      #pragma unroll
      for (int i = 0; i < 4; ++i) s[i][j] = __expf(s[i][j] - mnew);
      ol[j]   *= corr;
      o[0][j] *= corr; o[1][j] *= corr; o[2][j] *= corr; o[3][j] *= corr;
      // C-layout -> LDS (A-layout round trip)
      #pragma unroll
      for (int i = 0; i < 4; ++i)
        Plds[wave][j + 8 * hi][i * 16 + ln] = (_Float16)s[i][j];
    }
    asm volatile("s_wait_dscnt 0" ::: "memory");

    Frag16 ap0, ap1;   // P as A-matrix, kv chunks {0..31},{32..63}
    ap0.h[0] = *(const v8h*)&Plds[wave][ln][hi * 8];
    ap0.h[1] = *(const v8h*)&Plds[wave][ln][16 + hi * 8];
    ap1.h[0] = *(const v8h*)&Plds[wave][ln][32 + hi * 8];
    ap1.h[1] = *(const v8h*)&Plds[wave][ln][48 + hi * 8];

    // ---- O += P * V ; l += P * ones ----
    #pragma unroll
    for (int d = 0; d < 4; ++d) {
      Frag16 b0, b1;
      const int r = d * 16 + ln;                 // dk column
      b0.h[0] = *(const v8h*)&Vt[cur][r][hi * 8];
      b0.h[1] = *(const v8h*)&Vt[cur][r][16 + hi * 8];
      b1.h[0] = *(const v8h*)&Vt[cur][r][32 + hi * 8];
      b1.h[1] = *(const v8h*)&Vt[cur][r][48 + hi * 8];
      o[d] = __builtin_amdgcn_wmma_f32_16x16x32_f16(
          false, ap0.v, false, b0.v, (short)0, o[d], false, false);
      o[d] = __builtin_amdgcn_wmma_f32_16x16x32_f16(
          false, ap1.v, false, b1.v, (short)0, o[d], false, false);
    }
    ol = __builtin_amdgcn_wmma_f32_16x16x32_f16(
        false, ap0.v, false, ones.v, (short)0, ol, false, false);
    ol = __builtin_amdgcn_wmma_f32_16x16x32_f16(
        false, ap1.v, false, ones.v, (short)0, ol, false, false);
  }

  // ---- normalize and store f16 [B, S, H*DK] ----
  #pragma unroll
  for (int j = 0; j < 8; ++j) {
    const float inv = 1.0f / ol[j];
    const int qrow = q0 + j + 8 * hi;
    #pragma unroll
    for (int d = 0; d < 4; ++d) {
      const int col = h * DK + d * 16 + ln;
      Out[((size_t)b * SEQ + qrow) * D_MODEL + col] = (_Float16)(o[d][j] * inv);
    }
  }
}

// ---------------------------------------------------------------------------
extern "C" void kernel_launch(void* const* d_in, const int* in_sizes, int n_in,
                              void* d_out, int out_size, void* d_ws, size_t ws_size,
                              hipStream_t stream) {
  (void)in_sizes; (void)n_in; (void)out_size; (void)ws_size;

  const float* query = (const float*)d_in[0];
  const float* key   = (const float*)d_in[1];
  const float* value = (const float*)d_in[2];
  const float* Wq    = (const float*)d_in[3];
  const float* bq    = (const float*)d_in[4];
  const float* Wk    = (const float*)d_in[5];
  const float* bk    = (const float*)d_in[6];
  const float* Wv    = (const float*)d_in[7];
  const float* bv    = (const float*)d_in[8];
  const float* Wo    = (const float*)d_in[9];
  const float* bo    = (const float*)d_in[10];

  const int Mtot = 2 * SEQ;                       // 4096
  const size_t planeBytes = (size_t)Mtot * D_MODEL * sizeof(_Float16); // 8 MB
  char* ws = (char*)d_ws;
  _Float16* q16 = (_Float16*)(ws);
  _Float16* k16 = (_Float16*)(ws + planeBytes);
  _Float16* v16 = (_Float16*)(ws + 2 * planeBytes);
  _Float16* a16 = (_Float16*)(ws + 3 * planeBytes);

  dim3 blk(256);
  dim3 gGemm(D_MODEL / 64, Mtot / 128);           // (16, 32)

  gemm_wmma<false, true><<<gGemm, blk, 0, stream>>>(query, Wq, bq, q16);
  gemm_wmma<false, true><<<gGemm, blk, 0, stream>>>(key,   Wk, bk, k16);
  gemm_wmma<false, true><<<gGemm, blk, 0, stream>>>(value, Wv, bv, v16);

  dim3 gAttn(SEQ / 128, HEADS, 2);                // (16, 16, 2)
  attn_wmma<<<gAttn, blk, 0, stream>>>(q16, k16, v16, a16);

  gemm_wmma<true, false><<<gGemm, blk, 0, stream>>>(a16, Wo, bo, d_out);
}
